// Explainer_GIN_39608188404459
// MI455X (gfx1250) — matrix operand order repaired and checked by reading, not compiled, not run on hardware
//
#include <hip/hip_runtime.h>
#include <hip/hip_bf16.h>
#include <math.h>

#define NN 50000
#define NE 1600000
#define FIN 128
#define DIM 64
#define NG 500

typedef __attribute__((ext_vector_type(16))) __bf16 v16bf;
typedef __attribute__((ext_vector_type(8)))  float  v8f;

// ---------------- utility kernels ----------------
__global__ void k_fill(float* p, float v, int n) {
    int i = blockIdx.x * blockDim.x + threadIdx.x;
    if (i < n) p[i] = v;
}

__global__ void k_fill_dev(float* p, const float* v, int n) {
    int i = blockIdx.x * blockDim.x + threadIdx.x;
    if (i < n) p[i] = v[0];
}

__global__ void k_add_cvt(const float* __restrict__ a, const float* __restrict__ b,
                          __bf16* __restrict__ o, int n) {
    int i = blockIdx.x * blockDim.x + threadIdx.x;
    if (i < n) o[i] = (__bf16)(a[i] + b[i]);
}

__global__ void k_add(float* __restrict__ a, const float* __restrict__ b, int n) {
    int i = blockIdx.x * blockDim.x + threadIdx.x;
    if (i < n) a[i] += b[i];
}

// Pack W[K,64] (f32 row-major) into WMMA B-fragment order (bf16):
// Wp[(((ks*4 + nt)*32 + lane)*16) + e] = W[ks*32 + (e>>3)*16 + (lane>>4)*8 + (e&7)][nt*16 + (lane&15)]
// so a wave's B fragment for (k-step, n-tile) is one contiguous, 32B-aligned v16bf.
__global__ void k_pack_w(const float* __restrict__ W, __bf16* __restrict__ Wp, int K) {
    int idx = blockIdx.x * blockDim.x + threadIdx.x;
    if (idx >= K * DIM) return;
    int e    = idx & 15;
    int lane = (idx >> 4) & 31;
    int nt   = (idx >> 9) & 3;
    int ks   = idx >> 11;
    int k = ks * 32 + ((e >> 3) << 4) + ((lane >> 4) << 3) + (e & 7);
    int n = nt * 16 + (lane & 15);
    Wp[idx] = (__bf16)W[(size_t)k * DIM + n];
}

// ---------------- edge scatter-add: agg[dst] += h[src] ----------------
__global__ void k_scatter_add(const float* __restrict__ h, const int* __restrict__ src,
                              const int* __restrict__ dst, float* __restrict__ agg,
                              int nE, int F) {
    int tid = blockIdx.x * blockDim.x + threadIdx.x;
    int chunks = F >> 2;
    int total = nE * chunks;
    if (tid >= total) return;
    int e = tid / chunks;
    int f = (tid - e * chunks) << 2;
    int s = src[e], d = dst[e];
    const float4 v = *reinterpret_cast<const float4*>(h + (size_t)s * F + f);
    float* a = agg + (size_t)d * F + f;
    atomicAdd(a + 0, v.x);
    atomicAdd(a + 1, v.y);
    atomicAdd(a + 2, v.z);
    atomicAdd(a + 3, v.w);
}

// ---------------- WMMA GEMM: out = act(A[M,K] @ W[K,64] + bias) ----------------
// One wave per pair of 16x16 C tiles (two M tiles, one N tile). bf16 A/B, f32 acc.
// A fragment = 2 x b128 loads (lane's 16 elements are two runs of 8 consecutive K).
// B fragment = 1 x 32B load from the pre-packed Wp.
template <int K, bool RELU, bool WF, bool WB>
__global__ void k_wmma_gemm(const __bf16* __restrict__ A, const __bf16* __restrict__ Wp,
                            const float* __restrict__ bias,
                            float* __restrict__ outf, __bf16* __restrict__ outb, int M) {
    const int lane  = threadIdx.x & 31;
    const int wave  = threadIdx.x >> 5;
    const int tilesM = M >> 4;
    const int pairM  = (tilesM + 1) >> 1;
    int task = blockIdx.x * (blockDim.x >> 5) + wave;
    if (task >= pairM * 4) return;                 // wave-uniform exit
    const int mp = task >> 2;
    const int nt = task & 3;
    const int mt0 = mp * 2;
    const int mt1 = mt0 + 1;
    const bool has1 = (mt1 < tilesM);              // wave-uniform

    const int hi   = lane >> 4;                    // lane half
    const int lnlo = lane & 15;
    const int ncol = nt * 16 + lnlo;

    union AU { v16bf v; uint4 q[2]; };
    const v16bf* Bf = reinterpret_cast<const v16bf*>(Wp);
    const float  bs = bias[ncol];

    auto run_tile = [&](int mt) -> v8f {
        const uint4* Arow = reinterpret_cast<const uint4*>(A + (size_t)(mt * 16 + lnlo) * K);
        v8f acc = {};
        #pragma unroll
        for (int ks = 0; ks < K / 32; ++ks) {
            AU af;
            af.q[0] = Arow[ks * 4 + hi];           // K = ks*32 + hi*8      .. +8
            af.q[1] = Arow[ks * 4 + 2 + hi];       // K = ks*32 + 16 + hi*8 .. +8
            v16bf bv = Bf[(ks * 4 + nt) * 32 + lane];
            acc = __builtin_amdgcn_wmma_f32_16x16x32_bf16(
                false, af.v, false, bv, (short)0, acc, false, false);
        }
        return acc;
    };

    auto store_tile = [&](v8f acc, int mt) {
        #pragma unroll
        for (int r = 0; r < 8; ++r) {              // C VGPR r: M = r + 8*hi
            int m = mt * 16 + r + hi * 8;
            float v = acc[r] + bs;
            if (RELU) v = v > 0.f ? v : 0.f;
            size_t idx = (size_t)m * DIM + ncol;
            if (WF) outf[idx] = v;
            if (WB) outb[idx] = (__bf16)v;
        }
    };

    v8f acc0 = run_tile(mt0);
    store_tile(acc0, mt0);
    if (has1) {
        v8f acc1 = run_tile(mt1);
        store_tile(acc1, mt1);
    }
}

// ---------------- node_score += h[n,:] . wm[0:64] ----------------
__global__ void k_score_accum(const float* __restrict__ h, const float* __restrict__ wm,
                              float* __restrict__ score, int n) {
    int i = blockIdx.x * blockDim.x + threadIdx.x;
    if (i >= n) return;
    const float4* hr = reinterpret_cast<const float4*>(h + (size_t)i * DIM);
    const float4* wr = reinterpret_cast<const float4*>(wm);
    float s = 0.f;
    #pragma unroll
    for (int j = 0; j < DIM / 4; ++j) {
        float4 a = hr[j], b = wr[j];
        s += a.x * b.x + a.y * b.y + a.z * b.z + a.w * b.w;
    }
    score[i] += s;
}

// ---------------- float atomic max (signed-max / unsigned-min bit trick) --------
__device__ __forceinline__ void atomicMaxF(float* addr, float val) {
    if (val >= 0.f)
        atomicMax(reinterpret_cast<int*>(addr), __float_as_int(val));
    else
        atomicMin(reinterpret_cast<unsigned int*>(addr), __float_as_uint(val));
}

// ---------------- edge softmax passes (segments = dst) ----------------
__global__ void k_edge_max(const float* __restrict__ sc, const int* __restrict__ src,
                           const int* __restrict__ dst, float* __restrict__ nmax, int nE) {
    int e = blockIdx.x * blockDim.x + threadIdx.x;
    if (e >= nE) return;
    int d = dst[e];
    atomicMaxF(&nmax[d], sc[src[e]] * sc[d]);
}

__global__ void k_edge_den(const float* __restrict__ sc, const float* __restrict__ nmax,
                           const int* __restrict__ src, const int* __restrict__ dst,
                           float* __restrict__ den, int nE) {
    int e = blockIdx.x * blockDim.x + threadIdx.x;
    if (e >= nE) return;
    int d = dst[e];
    float ex = __expf(sc[src[e]] * sc[d] - nmax[d]);
    atomicAdd(&den[d], ex);
}

__global__ void k_edge_new(const float* __restrict__ sc, const float* __restrict__ nmax,
                           const float* __restrict__ den, const int* __restrict__ src,
                           const int* __restrict__ dst, float* __restrict__ nnew, int nE) {
    int e = blockIdx.x * blockDim.x + threadIdx.x;
    if (e >= nE) return;
    int d = dst[e];
    float ss = sc[src[e]];
    float prop = __expf(ss * sc[d] - nmax[d]) / den[d];
    atomicAdd(&nnew[d], ss * prop);
}

// ---------------- graph softmax passes (segments = batch) ----------------
__global__ void k_gmax(const float* __restrict__ ns, const int* __restrict__ batch,
                       float* __restrict__ gmax, int n) {
    int i = blockIdx.x * blockDim.x + threadIdx.x;
    if (i >= n) return;
    atomicMaxF(&gmax[batch[i]], ns[i]);
}

__global__ void k_gexp(const float* __restrict__ ns, const float* __restrict__ gmax,
                       const int* __restrict__ batch, float* __restrict__ gden,
                       float* __restrict__ oute, int n) {
    int i = blockIdx.x * blockDim.x + threadIdx.x;
    if (i >= n) return;
    int b = batch[i];
    float ex = __expf(ns[i] - gmax[b]);
    atomicAdd(&gden[b], ex);
    oute[i] = ex;
}

__global__ void k_gnorm(float* __restrict__ out, const float* __restrict__ gden,
                        const int* __restrict__ batch, int n) {
    int i = blockIdx.x * blockDim.x + threadIdx.x;
    if (i < n) out[i] /= gden[batch[i]];
}

// ---------------- host orchestration ----------------
static inline int nb(long long n, int t) { return (int)((n + t - 1) / t); }

extern "C" void kernel_launch(void* const* d_in, const int* in_sizes, int n_in,
                              void* d_out, int out_size, void* d_ws, size_t ws_size,
                              hipStream_t stream) {
    (void)in_sizes; (void)n_in; (void)out_size; (void)ws_size;

    const float* x     = (const float*)d_in[0];
    const int*   ei    = (const int*)d_in[1];
    const int*   src   = ei;             // edge_index[0]
    const int*   dst   = ei + NE;        // edge_index[1]
    const int*   batch = (const int*)d_in[2];
    const float* W1[3] = {(const float*)d_in[3],  (const float*)d_in[7],  (const float*)d_in[11]};
    const float* b1[3] = {(const float*)d_in[4],  (const float*)d_in[8],  (const float*)d_in[12]};
    const float* W2[3] = {(const float*)d_in[5],  (const float*)d_in[9],  (const float*)d_in[13]};
    const float* b2[3] = {(const float*)d_in[6],  (const float*)d_in[10], (const float*)d_in[14]};
    const float* Wm    = (const float*)d_in[15];
    const float* bm    = (const float*)d_in[16];
    float* out = (float*)d_out;

    // workspace carve-up (256B aligned)
    char* ws = (char*)d_ws;
    size_t off = 0;
    auto carve = [&](size_t bytes) -> char* {
        char* p = ws + off;
        off = (off + bytes + 255) & ~(size_t)255;
        return p;
    };
    float*  agg  = (float*) carve((size_t)NN * FIN * 4);
    __bf16* abf  = (__bf16*)carve((size_t)NN * FIN * 2);
    __bf16* tbf  = (__bf16*)carve((size_t)NN * DIM * 2);
    float*  hbuf = (float*) carve((size_t)NN * DIM * 4);
    __bf16* w1p[3]; __bf16* w2p[3];
    w1p[0] = (__bf16*)carve((size_t)FIN * DIM * 2);
    w2p[0] = (__bf16*)carve((size_t)DIM * DIM * 2);
    for (int i = 1; i < 3; ++i) {
        w1p[i] = (__bf16*)carve((size_t)DIM * DIM * 2);
        w2p[i] = (__bf16*)carve((size_t)DIM * DIM * 2);
    }
    float* nscore = (float*)carve((size_t)NN * 4);
    float* nmax   = (float*)carve((size_t)NN * 4);
    float* nden   = (float*)carve((size_t)NN * 4);
    float* nnew   = (float*)carve((size_t)NN * 4);
    float* gmax   = (float*)carve((size_t)NG * 4);
    float* gden   = (float*)carve((size_t)NG * 4);

    const int T = 256;

    // pack weights into bf16 B-fragment order
    k_pack_w<<<nb(FIN * DIM, T), T, 0, stream>>>(W1[0], w1p[0], FIN);
    k_pack_w<<<nb(DIM * DIM, T), T, 0, stream>>>(W2[0], w2p[0], DIM);
    for (int i = 1; i < 3; ++i) {
        k_pack_w<<<nb(DIM * DIM, T), T, 0, stream>>>(W1[i], w1p[i], DIM);
        k_pack_w<<<nb(DIM * DIM, T), T, 0, stream>>>(W2[i], w2p[i], DIM);
    }

    // node_score = bm (final-linear bias, added once)
    k_fill_dev<<<nb(NN, T), T, 0, stream>>>(nscore, bm, NN);

    const float* h = x;
    int F = FIN;
    const int tilesM = NN / 16;                       // 3125
    const int tasks  = ((tilesM + 1) / 2) * (DIM / 16);
    const int gemmBlocks = nb(tasks, 8);              // 8 waves/block

    for (int i = 0; i < 3; ++i) {
        // agg = segment_sum(h[src], dst)
        k_fill<<<nb((long long)NN * F, T), T, 0, stream>>>(agg, 0.f, NN * F);
        k_scatter_add<<<nb((long long)NE * (F / 4), T), T, 0, stream>>>(h, src, dst, agg, NE, F);
        // A = bf16(h + agg)
        k_add_cvt<<<nb((long long)NN * F, T), T, 0, stream>>>(h, agg, abf, NN * F);
        // t = relu(A @ W1 + b1) -> bf16
        if (F == FIN)
            k_wmma_gemm<FIN, true, false, true><<<gemmBlocks, T, 0, stream>>>(
                abf, w1p[i], b1[i], nullptr, tbf, NN);
        else
            k_wmma_gemm<DIM, true, false, true><<<gemmBlocks, T, 0, stream>>>(
                abf, w1p[i], b1[i], nullptr, tbf, NN);
        // h = t @ W2 + b2 (relu on layers 0,1) -> f32
        if (i < 2)
            k_wmma_gemm<DIM, true, true, false><<<gemmBlocks, T, 0, stream>>>(
                tbf, w2p[i], b2[i], hbuf, nullptr, NN);
        else
            k_wmma_gemm<DIM, false, true, false><<<gemmBlocks, T, 0, stream>>>(
                tbf, w2p[i], b2[i], hbuf, nullptr, NN);
        // node_score += h @ Wm[i*64:(i+1)*64]
        k_score_accum<<<nb(NN, T), T, 0, stream>>>(hbuf, Wm + i * DIM, nscore, NN);
        h = hbuf;
        F = DIM;
    }

    // edge softmax over dst + weighted scatter
    k_fill<<<nb(NN, T), T, 0, stream>>>(nmax, -INFINITY, NN);
    k_fill<<<nb(NN, T), T, 0, stream>>>(nden, 0.f, NN);
    k_fill<<<nb(NN, T), T, 0, stream>>>(nnew, 0.f, NN);
    k_edge_max<<<nb(NE, T), T, 0, stream>>>(nscore, src, dst, nmax, NE);
    k_edge_den<<<nb(NE, T), T, 0, stream>>>(nscore, nmax, src, dst, nden, NE);
    k_edge_new<<<nb(NE, T), T, 0, stream>>>(nscore, nmax, nden, src, dst, nnew, NE);
    k_add<<<nb(NN, T), T, 0, stream>>>(nscore, nnew, NN);

    // graph softmax over batch
    k_fill<<<nb(NG, T), T, 0, stream>>>(gmax, -INFINITY, NG);
    k_fill<<<nb(NG, T), T, 0, stream>>>(gden, 0.f, NG);
    k_gmax<<<nb(NN, T), T, 0, stream>>>(nscore, batch, gmax, NN);
    k_gexp<<<nb(NN, T), T, 0, stream>>>(nscore, gmax, batch, gden, out, NN);
    k_gnorm<<<nb(NN, T), T, 0, stream>>>(out, gden, batch, NN);
}
